// LSTM_5703716569095
// MI455X (gfx1250) — matrix operand order repaired and checked by reading, not compiled, not run on hardware
//
#include <hip/hip_runtime.h>

// Problem constants (from reference): T=128, B=64, I=H=1024, L=8
#define T_STEPS 128
#define B_SZ    64
#define I_SZ    1024
#define H_SZ    1024
#define L_SZ    8

typedef __attribute__((ext_vector_type(16))) __bf16 v16bf;
typedef __attribute__((ext_vector_type(8)))  __bf16 v8bf;
typedef __attribute__((ext_vector_type(8)))  float  v8f;

__device__ __forceinline__ unsigned short f32_to_bf16_rne(float f) {
  union { float f; unsigned int u; } cv; cv.f = f;
  unsigned int u = cv.u;
  unsigned int r = (u + 0x7FFFu + ((u >> 16) & 1u)) >> 16;
  return (unsigned short)r;
}

__device__ __forceinline__ float sigmoid_fast(float x) {
  return 1.0f / (1.0f + __expf(-x));
}
__device__ __forceinline__ float tanh_fast(float x) {
  float e = __expf(-2.0f * x);
  return (1.0f - e) / (1.0f + e);
}

// A fragment (16x32 bf16): lane = row m; elems 0..15 -> K = khalf*16 + 0..15.
__device__ __forceinline__ v16bf ld_a(const unsigned short* Ap, int kt, int khalf) {
  v8bf lo = *(const v8bf*)(Ap + kt + khalf * 16);
  v8bf hi = *(const v8bf*)(Ap + kt + khalf * 16 + 8);
  return __builtin_shufflevector(lo, hi, 0,1,2,3,4,5,6,7,8,9,10,11,12,13,14,15);
}
// B fragment (32x16 bf16): lane = col n; elems 0..15 -> K = khalf*16 + 0..15 (pre-transposed W^T[n][k]).
__device__ __forceinline__ v16bf ld_b(const unsigned short* Bp, int kb) {
  v8bf lo = *(const v8bf*)(Bp + kb);
  v8bf hi = *(const v8bf*)(Bp + kb + 8);
  return __builtin_shufflevector(lo, hi, 0,1,2,3,4,5,6,7,8,9,10,11,12,13,14,15);
}

// ---- Prep kernels -----------------------------------------------------------

// in: nmat matrices of [1024 x 1024] fp32 (k-major rows), out: bf16 [1024(n) x 1024(k)]
__global__ __launch_bounds__(256) void k_transpose_bf16(
    const float* __restrict__ in, unsigned short* __restrict__ out, int nmat) {
  size_t idx = (size_t)blockIdx.x * 256 + threadIdx.x;
  size_t total = (size_t)nmat << 20;
  if (idx >= total) return;
  int m   = (int)(idx >> 20);
  int rem = (int)(idx & 1048575u);
  int k = rem >> 10, n = rem & 1023;
  out[((size_t)m << 20) + ((size_t)n << 10) + (size_t)k] = f32_to_bf16_rne(in[idx]);
}

__global__ __launch_bounds__(256) void k_f32_to_bf16(
    const float* __restrict__ in, unsigned short* __restrict__ out, int n) {
  int i = blockIdx.x * 256 + threadIdx.x;
  if (i < n) out[i] = f32_to_bf16_rne(in[i]);
}

__global__ __launch_bounds__(256) void k_bias_add(
    const float* __restrict__ a, const float* __restrict__ b,
    float* __restrict__ o, int n) {
  int i = blockIdx.x * 256 + threadIdx.x;
  if (i < n) o[i] = a[i] + b[i];
}

__global__ __launch_bounds__(256) void k_zero16(float4* __restrict__ p, int n16) {
  int i = blockIdx.x * 256 + threadIdx.x;
  if (i < n16) p[i] = make_float4(0.f, 0.f, 0.f, 0.f);
}

// ---- Fused LSTM cell: gates GEMM (bf16 WMMA, split-K, depth-2 SW pipeline) --
//
// Output tile space: 4 b-tiles (B=64) x 64 n-tiles (H=1024) = 256 blocks.
// Each block: 4 waves; wave w reduces K range [w*512, w*512+512) of the
// concatenated K = [x(1024) | h(1024)]. Each wave carries 4 gate accumulators
// sharing one A fragment. Fragments rotate through 3 register stages with a
// depth-2 prefetch. The prologue's 20 loads are declared as their own sched
// group so the per-iteration [10 VMEM, 4 WMMA] groups bind to the k+2
// prefetch, keeping ~20 loads in flight across every WMMA block.
__global__ __launch_bounds__(128, 1) void k_lstm_cell(
    const unsigned short* __restrict__ xin,   // [64,1024] bf16 (input from below / x_t)
    const unsigned short* __restrict__ hin,   // [64,1024] bf16 (this layer's h_{t-1})
    const unsigned short* __restrict__ WihT,  // [4,1024(n),1024(k)] bf16
    const unsigned short* __restrict__ WhhT,  // [4,1024(n),1024(k)] bf16
    const float*          __restrict__ bias,  // [4,1024] fp32 (bih+bhh)
    float*                __restrict__ cst,   // [64,1024] fp32, in-place
    unsigned short*       __restrict__ hout,  // [64,1024] bf16 (double-buffered)
    float*                __restrict__ hout_f32) // optional fp32 mirror (final output)
{
  __shared__ float red[4][4][32][8];  // [wave][gate][lane][row]

  const int tid  = threadIdx.x;
  const int wave = tid >> 5;
  const int lane = tid & 31;
  const int btile = (blockIdx.x & 3) << 4;   // 0,16,32,48
  const int ntile = (blockIdx.x >> 2) << 4;  // 0..1008

  // split-K: waves 0,1 -> x . Wih ; waves 2,3 -> h . Whh
  const unsigned short* Asrc = (wave < 2) ? xin  : hin;
  const unsigned short* Wsrc = (wave < 2) ? WihT : WhhT;
  const int kbase = (wave & 1) * 512;

  const int mrow  = btile + (lane & 15);
  const int ncol  = ntile + (lane & 15);
  const int khalf = lane >> 4;

  const unsigned short* Ap  = Asrc + (size_t)mrow * 1024 + kbase;
  const unsigned short* Bp0 = Wsrc + (size_t)0 * 1048576 + (size_t)ncol * 1024 + kbase;
  const unsigned short* Bp1 = Wsrc + (size_t)1 * 1048576 + (size_t)ncol * 1024 + kbase;
  const unsigned short* Bp2 = Wsrc + (size_t)2 * 1048576 + (size_t)ncol * 1024 + kbase;
  const unsigned short* Bp3 = Wsrc + (size_t)3 * 1048576 + (size_t)ncol * 1024 + kbase;

  v8f acc0 = {}, acc1 = {}, acc2 = {}, acc3 = {};

  // 3 rotating fragment stages for a depth-2 prefetch (no WAR between a
  // prefetch store and the WMMA consuming the 2-steps-older fragment).
  v16bf Af[3], B0f[3], B1f[3], B2f[3], B3f[3];

  auto LOAD_STAGE = [&](int s, int kt) {
    Af[s] = ld_a(Ap, kt, khalf);
    const int kbn = kt + khalf * 16;
    B0f[s] = ld_b(Bp0, kbn);
    B1f[s] = ld_b(Bp1, kbn);
    B2f[s] = ld_b(Bp2, kbn);
    B3f[s] = ld_b(Bp3, kbn);
  };

  // Prologue: K-steps 0 and 1 in flight. Declared as an explicit sched group
  // so the in-loop VMEM groups bind to the k+2 prefetch loads, not these.
  LOAD_STAGE(0, 0);
  LOAD_STAGE(1, 32);
  __builtin_amdgcn_sched_group_barrier(0x020, 20, 0); // prologue VMEM reads

#pragma unroll
  for (int i = 0; i < 16; ++i) {
    if (i + 2 < 16) LOAD_STAGE((i + 2) % 3, (i + 2) * 32);

    const int cu = i % 3;
    acc0 = __builtin_amdgcn_wmma_f32_16x16x32_bf16(false, Af[cu], false, B0f[cu], (short)0, acc0, false, false);
    acc1 = __builtin_amdgcn_wmma_f32_16x16x32_bf16(false, Af[cu], false, B1f[cu], (short)0, acc1, false, false);
    acc2 = __builtin_amdgcn_wmma_f32_16x16x32_bf16(false, Af[cu], false, B2f[cu], (short)0, acc2, false, false);
    acc3 = __builtin_amdgcn_wmma_f32_16x16x32_bf16(false, Af[cu], false, B3f[cu], (short)0, acc3, false, false);

    // Schedule shape per K-step: 10 VMEM reads (prefetch k+2), then 4 WMMAs (k).
    __builtin_amdgcn_sched_group_barrier(0x020, 10, 0); // VMEM read group
    __builtin_amdgcn_sched_group_barrier(0x008, 4, 0);  // MFMA/WMMA group
  }

  // Cross-wave split-K reduction through LDS.
#pragma unroll
  for (int r = 0; r < 8; ++r) {
    red[wave][0][lane][r] = acc0[r];
    red[wave][1][lane][r] = acc1[r];
    red[wave][2][lane][r] = acc2[r];
    red[wave][3][lane][r] = acc3[r];
  }
  __syncthreads();

  // Pointwise LSTM epilogue: 256 tile elements over 128 threads (2 each).
  // C/D layout (ISA 7.12.2): elem r of lane l -> m = r + (l>>4)*8, n = l&15.
#pragma unroll
  for (int e2 = 0; e2 < 2; ++e2) {
    const int e = tid * 2 + e2;
    const int m = e >> 4, n = e & 15;
    const int lc = n + ((m & 8) << 1);  // lane holding (m,n)
    const int r  = m & 7;               // accumulator row

    float g0 = bias[0 * 1024 + ntile + n];
    float g1 = bias[1 * 1024 + ntile + n];
    float g2 = bias[2 * 1024 + ntile + n];
    float g3 = bias[3 * 1024 + ntile + n];
#pragma unroll
    for (int w = 0; w < 4; ++w) {
      g0 += red[w][0][lc][r];
      g1 += red[w][1][lc][r];
      g2 += red[w][2][lc][r];
      g3 += red[w][3][lc][r];
    }

    const int idx = (btile + m) * 1024 + ntile + n;
    const float ig = sigmoid_fast(g0);
    const float fg = sigmoid_fast(g1);
    const float cg = tanh_fast(g2);
    const float og = sigmoid_fast(g3);
    const float cnew = fg * cst[idx] + ig * cg;
    const float hnew = og * tanh_fast(cnew);
    cst[idx]  = cnew;
    hout[idx] = f32_to_bf16_rne(hnew);
    if (hout_f32) hout_f32[idx] = hnew;
  }
}

// ---- Host orchestration -----------------------------------------------------

extern "C" void kernel_launch(void* const* d_in, const int* in_sizes, int n_in,
                              void* d_out, int out_size, void* d_ws, size_t ws_size,
                              hipStream_t stream) {
  (void)in_sizes; (void)n_in; (void)out_size; (void)ws_size;

  const float* inputs = (const float*)d_in[0];  // [T,B,I]
  const float* Wih    = (const float*)d_in[1];  // [L,4,I,H]
  const float* Whh    = (const float*)d_in[2];  // [L,4,H,H]
  const float* bih    = (const float*)d_in[3];  // [L,4,H]
  const float* bhh    = (const float*)d_in[4];  // [L,4,H]

  // Workspace layout (all offsets 256B-aligned by construction):
  char* ws = (char*)d_ws;
  size_t off = 0;
  unsigned short* WihT = (unsigned short*)(ws + off); off += (size_t)L_SZ * 4 * 1024 * 1024 * 2; // 64 MB
  unsigned short* WhhT = (unsigned short*)(ws + off); off += (size_t)L_SZ * 4 * 1024 * 1024 * 2; // 64 MB
  unsigned short* xbf  = (unsigned short*)(ws + off); off += (size_t)T_STEPS * B_SZ * I_SZ * 2;  // 16 MB
  unsigned short* hbuf = (unsigned short*)(ws + off); off += (size_t)2 * L_SZ * B_SZ * H_SZ * 2; // 2 MB (double buffer)
  float* cst           = (float*)(ws + off);          off += (size_t)L_SZ * B_SZ * H_SZ * 4;     // 2 MB
  float* biasw         = (float*)(ws + off);          off += (size_t)L_SZ * 4 * H_SZ * 4;        // 128 KB

  // --- one-time prep (re-run every call for determinism) ---
  {
    const int nmat = L_SZ * 4;
    const size_t total = (size_t)nmat << 20;
    const int blocks = (int)((total + 255) / 256);
    k_transpose_bf16<<<blocks, 256, 0, stream>>>(Wih, WihT, nmat);
    k_transpose_bf16<<<blocks, 256, 0, stream>>>(Whh, WhhT, nmat);
  }
  {
    const int n = T_STEPS * B_SZ * I_SZ;
    k_f32_to_bf16<<<(n + 255) / 256, 256, 0, stream>>>(inputs, xbf, n);
  }
  {
    const int n = L_SZ * 4 * H_SZ;
    k_bias_add<<<(n + 255) / 256, 256, 0, stream>>>(bih, bhh, biasw, n);
  }
  {
    // hbuf (bf16, 2MB) and cst (fp32, 2MB) are contiguous: zero both in one pass.
    const int n16 = (2 * L_SZ * B_SZ * H_SZ * 2 + L_SZ * B_SZ * H_SZ * 4) / 16;
    k_zero16<<<(n16 + 255) / 256, 256, 0, stream>>>((float4*)hbuf, n16);
  }

  // --- recurrent sweep: T x L fused cell kernels, serialized on stream ---
  const size_t LBH = (size_t)L_SZ * B_SZ * H_SZ;
  for (int t = 0; t < T_STEPS; ++t) {
    const size_t cur = (size_t)(t & 1), nxt = cur ^ 1;
    for (int j = 0; j < L_SZ; ++j) {
      const unsigned short* xin =
          (j == 0) ? (xbf + (size_t)t * B_SZ * I_SZ)
                   : (hbuf + nxt * LBH + (size_t)(j - 1) * B_SZ * H_SZ);
      const unsigned short* hin = hbuf + cur * LBH + (size_t)j * B_SZ * H_SZ;
      unsigned short* hO        = hbuf + nxt * LBH + (size_t)j * B_SZ * H_SZ;
      float* ho32 = (t == T_STEPS - 1 && j == L_SZ - 1) ? (float*)d_out : nullptr;

      k_lstm_cell<<<256, 128, 0, stream>>>(
          xin, hin,
          WihT + (size_t)j * 4 * 1024 * 1024,
          WhhT + (size_t)j * 4 * 1024 * 1024,
          biasw + (size_t)j * 4 * H_SZ,
          cst + (size_t)j * B_SZ * H_SZ,
          hO, ho32);
    }
  }
}
